// LSTM_39230231281908
// MI455X (gfx1250) — matrix operand order repaired and checked by reading, not compiled, not run on hardware
//
#include <hip/hip_runtime.h>
#include <hip/hip_bf16.h>

// ---------------------------------------------------------------------------
// LSTM-like scan:  xw = x@W + b  (big GEMM, full chip, WMMA f16->f32)
//                  then per-step:  a = xw_t + h @ recur_w ; gates ; emit s
// B=64, T=1024, I=H=512, 4H=2048.
// ---------------------------------------------------------------------------

typedef __attribute__((ext_vector_type(16))) _Float16 v16h;
typedef __attribute__((ext_vector_type(8)))  float    v8f;
typedef __attribute__((ext_vector_type(4)))  float    f32x4;

#define BB   64
#define TT   1024
#define KK   512      // I == H
#define NN   2048     // 4H

// ---- WMMA fragment loaders -------------------------------------------------
// A fragment (16x32 f16, MxK) from row-major f16 memory.
// Layout: lanes 0-15 hold M=lane, K = {0..15}; lanes 16-31 hold
// M=lane-16, K = {16..31}.  Each half is two contiguous 16B loads.
__device__ __forceinline__ v16h load_a_frag(const _Float16* __restrict__ base,
                                            int ld, int m0, int k0, int lane) {
  const _Float16* p = base + (m0 + (lane & 15)) * ld + k0 + 16 * (lane >> 4);
  v16h r;
  *reinterpret_cast<f32x4*>(&r)       = *reinterpret_cast<const f32x4*>(p);
  *(reinterpret_cast<f32x4*>(&r) + 1) = *reinterpret_cast<const f32x4*>(p + 8);
  return r;
}

// B fragment (32x16 f16, KxN) from pre-swizzled weights: element (k,n) lives
// at lane = (n&15) + 16*(k>>4), half = k&15.  Swizzler stores each lane's 16
// halves contiguously -> one 32B load per lane.
__device__ __forceinline__ v16h load_b_frag(const _Float16* __restrict__ swz,
                                            int nt, int kc, int lane) {
  const _Float16* p = swz + (((nt * 16 + kc) * 32) + lane) * 16;
  v16h r;
  *reinterpret_cast<f32x4*>(&r)       = *reinterpret_cast<const f32x4*>(p);
  *(reinterpret_cast<f32x4*>(&r) + 1) = *reinterpret_cast<const f32x4*>(p + 8);
  return r;
}

__device__ __forceinline__ float sigmoidf_fast(float x) {
  return 1.f / (1.f + __expf(-x));
}

// ---- Phase 0a: x f32 -> f16 (row-major, becomes WMMA A operand) ------------
__global__ void __launch_bounds__(256) k_convert_x(const float* __restrict__ x,
                                                   _Float16* __restrict__ xh,
                                                   int n) {
  for (int i = blockIdx.x * blockDim.x + threadIdx.x; i < n;
       i += gridDim.x * blockDim.x)
    xh[i] = (_Float16)x[i];
}

// ---- Phase 0b: swizzle W and recur_w into B-fragment order -----------------
__global__ void __launch_bounds__(256) k_swizzle_w(const float* __restrict__ W,
                                                   const float* __restrict__ R,
                                                   _Float16* __restrict__ Wz,
                                                   _Float16* __restrict__ Rz) {
  const int total = KK * NN;
  for (int i = blockIdx.x * blockDim.x + threadIdx.x; i < total;
       i += gridDim.x * blockDim.x) {
    int k = i >> 11;           // row (K)
    int n = i & (NN - 1);      // col (N)
    int nt = n >> 4, kc = k >> 5;
    int lane = (n & 15) | (((k >> 4) & 1) << 4);
    int e    = k & 15;
    int d = (((nt * 16 + kc) * 32) + lane) * 16 + e;
    Wz[d] = (_Float16)W[i];
    Rz[d] = (_Float16)R[i];
  }
}

// ---- Phase 0c: zero h double-buffer + sync counters (every launch) ---------
__global__ void __launch_bounds__(256) k_init(_Float16* __restrict__ hbuf,
                                              unsigned* __restrict__ counters) {
  int i = blockIdx.x * blockDim.x + threadIdx.x;
  const int nh = 2 * BB * KK;
  for (int j = i; j < nh; j += gridDim.x * blockDim.x)
    hbuf[j] = (_Float16)0.f;
  if (i < 128) counters[i] = 0u;
}

// ---- Phase 1: xw = x @ W + b  (M=65536, K=512, N=2048), out f16 ------------
// Block = 8 waves, computes a 64x128 tile; wave w -> M-tile (w&3), N-tiles
// (w>>2)*4 .. +3.  Output written transposed to [t][b][4H] for the scan.
__global__ void __launch_bounds__(256) k_gemm_xw(const _Float16* __restrict__ xh,
                                                 const _Float16* __restrict__ Wz,
                                                 const float* __restrict__ bias,
                                                 _Float16* __restrict__ xw) {
  const int lane = threadIdx.x & 31;
  const int w    = threadIdx.x >> 5;
  const int nblk = blockIdx.x & 15;   // 16 N-blocks of 128 cols
  const int mblk = blockIdx.x >> 4;   // 1024 M-blocks of 64 rows
  const int mi     = w & 3;
  const int njgrp  = (w >> 2) * 4;
  const int m0     = mblk * 64 + mi * 16;

  v8f acc[4] = {};
#pragma unroll 4
  for (int kc = 0; kc < KK / 32; ++kc) {
    v16h a = load_a_frag(xh, KK, m0, kc * 32, lane);
#pragma unroll
    for (int q = 0; q < 4; ++q) {
      v16h b = load_b_frag(Wz, nblk * 8 + njgrp + q, kc, lane);
      acc[q] = __builtin_amdgcn_wmma_f32_16x16x32_f16(
          false, a, false, b, (short)0, acc[q], false, false);
    }
  }
#pragma unroll
  for (int q = 0; q < 4; ++q) {
    const int col = (nblk * 8 + njgrp + q) * 16 + (lane & 15);
    const float bv = bias[col];
#pragma unroll
    for (int r = 0; r < 8; ++r) {
      int m  = m0 + r + 8 * (lane >> 4);   // m = b*T + t
      int t  = m & (TT - 1);
      int bb = m >> 10;
      xw[(long)(t * BB + bb) * NN + col] = (_Float16)(acc[q][r] + bv);
    }
  }
}

// ---- Phase 2: sequential scan ----------------------------------------------
// 4 independent row-groups (16 batch rows each) x 16 column-blocks (32 j each).
// Per step each block: a[16x128] = h[16x512] @ recur_w[:,4 coupled strips],
// stage to LDS, gates, write s to out and new h (f16, double buffered), then
// 16-way per-group atomic barrier.
__global__ void __launch_bounds__(256, 1) k_scan(const _Float16* __restrict__ xw,
                                                 const _Float16* __restrict__ Rz,
                                                 _Float16* __restrict__ hbuf,
                                                 unsigned* __restrict__ counters,
                                                 float* __restrict__ out) {
  const int g    = blockIdx.x >> 4;    // row group: batches 16g..16g+15
  const int cb   = blockIdx.x & 15;    // j0 = 32*cb
  const int tid  = threadIdx.x;
  const int lane = tid & 31;
  const int w    = tid >> 5;           // 8 waves -> 8 N-tiles
  __shared__ float a_lds[16][128];

  unsigned* cnt = counters + g * 32;   // 128B apart per group
  _Float16* h0 = hbuf;
  _Float16* h1 = hbuf + BB * KK;

  const int strip = w >> 1;            // gate index 0..3
  const int nt = (strip * 512 + cb * 32 + (w & 1) * 16) >> 4;  // swizzled tile

  float s0 = 0.f, s1 = 0.f;            // cell state, fixed (b,j) per thread

  for (int t = 0; t < TT; ++t) {
    const _Float16* __restrict__ hcur = (t & 1) ? h1 : h0;
    _Float16* __restrict__ hnext      = (t & 1) ? h0 : h1;

    // --- recurrent GEMM: one 16x16 tile per wave, K=512 -------------------
    v8f acc = {};
#pragma unroll 4
    for (int kc = 0; kc < KK / 32; ++kc) {
      v16h a = load_a_frag(hcur, KK, g * 16, kc * 32, lane);
      v16h b = load_b_frag(Rz, nt, kc, lane);
      acc = __builtin_amdgcn_wmma_f32_16x16x32_f16(
          false, a, false, b, (short)0, acc, false, false);
    }
#pragma unroll
    for (int r = 0; r < 8; ++r)
      a_lds[r + 8 * (lane >> 4)][w * 16 + (lane & 15)] = acc[r];

    // prefetch next step's xw slice while tiles drain (global_prefetch_b8)
    if (t + 1 < TT) {
      int prow = tid & 15, pst = tid >> 6;
      __builtin_prefetch(
          xw + (long)((t + 1) * BB + g * 16 + prow) * NN + pst * 512 + cb * 32,
          0, 1);
    }
    __syncthreads();

    // --- gates: 512 (row,j) pairs, 2 per thread ---------------------------
#pragma unroll
    for (int half = 0; half < 2; ++half) {
      const int p   = tid + half * 256;
      const int row = p >> 5;                // 0..15
      const int jl  = p & 31;
      const int bg  = g * 16 + row;          // batch index
      const int jg  = cb * 32 + jl;          // 0..511
      const _Float16* xr = xw + (long)(t * BB + bg) * NN;
      float a1 = a_lds[row][0 * 32 + jl] + (float)xr[0 * 512 + jg];
      float a2 = a_lds[row][1 * 32 + jl] + (float)xr[1 * 512 + jg];
      float a3 = a_lds[row][2 * 32 + jl] + (float)xr[2 * 512 + jg];
      float a4 = a_lds[row][3 * 32 + jl] + (float)xr[3 * 512 + jg];
      float& s = half ? s1 : s0;
      s = sigmoidf_fast(a1) * s + sigmoidf_fast(a2) * tanhf(a3);
      float h = tanhf(s) * sigmoidf_fast(a4);
      out[((long)bg * TT + t) * KK + jg] = s;       // reference emits s_t
      hnext[bg * KK + jg] = (_Float16)h;
    }

    // --- per-group barrier: all 16 column-blocks finished this step -------
    __threadfence();
    __syncthreads();
    if (tid == 0) {
      __hip_atomic_fetch_add(cnt, 1u, __ATOMIC_ACQ_REL, __HIP_MEMORY_SCOPE_AGENT);
      const unsigned target = 16u * (unsigned)(t + 1);
      while (__hip_atomic_load(cnt, __ATOMIC_ACQUIRE, __HIP_MEMORY_SCOPE_AGENT) <
             target)
        __builtin_amdgcn_s_sleep(2);
    }
    __syncthreads();
  }
}

// ---------------------------------------------------------------------------
extern "C" void kernel_launch(void* const* d_in, const int* in_sizes, int n_in,
                              void* d_out, int out_size, void* d_ws,
                              size_t ws_size, hipStream_t stream) {
  (void)in_sizes; (void)n_in; (void)out_size; (void)ws_size;
  const float* x  = (const float*)d_in[0];   // [64,1024,512]
  const float* W  = (const float*)d_in[1];   // [512,2048]
  const float* R  = (const float*)d_in[2];   // [512,2048]
  const float* bv = (const float*)d_in[3];   // [1,2048]
  float* out = (float*)d_out;                // [64,1024,512]

  char* ws = (char*)d_ws;
  size_t off = 0;
  _Float16* xh = (_Float16*)(ws + off); off += (size_t)BB * TT * KK * 2;  // 64 MB
  _Float16* Wz = (_Float16*)(ws + off); off += (size_t)KK * NN * 2;       //  2 MB
  _Float16* Rz = (_Float16*)(ws + off); off += (size_t)KK * NN * 2;       //  2 MB
  _Float16* xw = (_Float16*)(ws + off); off += (size_t)TT * BB * NN * 2;  // 256 MB
  _Float16* hb = (_Float16*)(ws + off); off += (size_t)2 * BB * KK * 2;   // 128 KB
  unsigned* cnt = (unsigned*)(ws + off); off += 512;

  k_convert_x<<<4096, 256, 0, stream>>>(x, xh, BB * TT * KK);
  k_swizzle_w<<<1024, 256, 0, stream>>>(W, R, Wz, Rz);
  k_init<<<64, 256, 0, stream>>>(hb, cnt);
  k_gemm_xw<<<(BB * TT / 64) * (NN / 128), 256, 0, stream>>>(xh, Wz, bv, xw);
  k_scan<<<64, 256, 0, stream>>>(xw, Rz, hb, cnt, out);
}
